// SHORTSIREN_F_26164940767687
// MI455X (gfx1250) — matrix operand order repaired and checked by reading, not compiled, not run on hardware
//
#include <hip/hip_runtime.h>

typedef _Float16 half_t;
typedef __attribute__((ext_vector_type(16))) _Float16 v16h;
typedef __attribute__((ext_vector_type(8)))  _Float16 v8h;
typedef __attribute__((ext_vector_type(8)))  float    v8f;

#define HID   256
#define C_IN  32
#define RES   32
#define ROWS_PER_WAVE  32            // two 16-row M-tiles per wave
#define ROWS_PER_BLOCK 256           // 8 waves

union V16 { v16h v; v8h h[2]; };

// ---------------------------------------------------------------------------
// Pack W (Nout x K, f32, row-major) into per-lane WMMA B-fragment layout:
// dst[((t*ktiles + kt)*32 + lane)*16 + j] = W[t*16 + (lane&15)][kt*32 + ((j>=8)?16:0) + ((lane>=16)?8:0) + (j&7)]
// Zero-pad outside (Nout, K) (used for the 4-wide final layer padded to 16).
// ---------------------------------------------------------------------------
__global__ void pack_w_kernel(const float* __restrict__ W, half_t* __restrict__ dst,
                              int Nout, int K, int ntiles, int ktiles) {
  int idx = blockIdx.x * 256 + threadIdx.x;
  int total = ntiles * ktiles * 512;
  if (idx >= total) return;
  int t    = idx / (ktiles * 512);
  int r    = idx % (ktiles * 512);
  int kt   = r / 512;
  int q    = r % 512;
  int lane = q / 16;
  int j    = q % 16;
  int n = t * 16 + (lane & 15);
  int k = kt * 32 + ((j >= 8) ? 16 : 0) + ((lane >= 16) ? 8 : 0) + (j & 7);
  float val = (n < Nout && k < K) ? W[(size_t)n * K + k] : 0.0f;
  dst[idx] = (half_t)val;
}

// ---------------------------------------------------------------------------
// Trilinear grid-sample, align_corners=False, border padding.
// feats[tid][c] (f16, row-major, 32 per point) == layer-0 A source.
// ---------------------------------------------------------------------------
__global__ void grid_sample_kernel(const float* __restrict__ points,
                                   const float* __restrict__ vol,
                                   half_t* __restrict__ feats,
                                   int total, int Nper) {
  int tid = blockIdx.x * 256 + threadIdx.x;
  if (tid >= total) return;
  int b = tid / Nper;
  const float inv_half_vox = 1.0f / 0.6f;   // grid = points / (VOXEL_LENGTH/2)
  float gx = points[(size_t)tid * 3 + 0] * inv_half_vox;
  float gy = points[(size_t)tid * 3 + 1] * inv_half_vox;
  float gz = points[(size_t)tid * 3 + 2] * inv_half_vox;
  float ix = ((gx + 1.0f) * RES - 1.0f) * 0.5f;
  float iy = ((gy + 1.0f) * RES - 1.0f) * 0.5f;
  float iz = ((gz + 1.0f) * RES - 1.0f) * 0.5f;
  float fx = floorf(ix), fy = floorf(iy), fz = floorf(iz);
  float wx = ix - fx, wy = iy - fy, wz = iz - fz;
  int x0 = (int)fminf(fmaxf(fx,        0.0f), (float)(RES - 1));
  int x1 = (int)fminf(fmaxf(fx + 1.0f, 0.0f), (float)(RES - 1));
  int y0 = (int)fminf(fmaxf(fy,        0.0f), (float)(RES - 1));
  int y1 = (int)fminf(fmaxf(fy + 1.0f, 0.0f), (float)(RES - 1));
  int z0 = (int)fminf(fmaxf(fz,        0.0f), (float)(RES - 1));
  int z1 = (int)fminf(fmaxf(fz + 1.0f, 0.0f), (float)(RES - 1));

  int i000 = (z0 * RES + y0) * RES + x0;
  int i001 = (z0 * RES + y0) * RES + x1;
  int i010 = (z0 * RES + y1) * RES + x0;
  int i011 = (z0 * RES + y1) * RES + x1;
  int i100 = (z1 * RES + y0) * RES + x0;
  int i101 = (z1 * RES + y0) * RES + x1;
  int i110 = (z1 * RES + y1) * RES + x0;
  int i111 = (z1 * RES + y1) * RES + x1;

  float w000 = (1 - wz) * (1 - wy) * (1 - wx);
  float w001 = (1 - wz) * (1 - wy) * wx;
  float w010 = (1 - wz) * wy * (1 - wx);
  float w011 = (1 - wz) * wy * wx;
  float w100 = wz * (1 - wy) * (1 - wx);
  float w101 = wz * (1 - wy) * wx;
  float w110 = wz * wy * (1 - wx);
  float w111 = wz * wy * wx;

  const float* vb = vol + (size_t)b * C_IN * (RES * RES * RES);
  half_t* fo = feats + (size_t)tid * C_IN;
#pragma unroll 4
  for (int c = 0; c < C_IN; ++c) {
    const float* vc = vb + (size_t)c * (RES * RES * RES);
    float r = vc[i000] * w000 + vc[i001] * w001 + vc[i010] * w010 + vc[i011] * w011
            + vc[i100] * w100 + vc[i101] * w101 + vc[i110] * w110 + vc[i111] * w111;
    fo[c] = (half_t)r;
  }
}

// ---------------------------------------------------------------------------
// Fused SIREN MLP with WMMA f32_16x16x32_f16, 2 M-tiles per wave for 2x
// register reuse of every B fragment.
// ---------------------------------------------------------------------------
__device__ __forceinline__ v8f wmma_step(v16h a, v16h b, v8f c) {
  return __builtin_amdgcn_wmma_f32_16x16x32_f16(false, a, false, b, (short)0, c, false, false);
}

__device__ __forceinline__ v16h load_afrag_lds(const half_t* chunk, int row, int kt, int kbase) {
  const half_t* ap = chunk + row * HID + kt * 32;
  V16 u;
  u.h[0] = *(const v8h*)(ap + kbase);
  u.h[1] = *(const v8h*)(ap + 16 + kbase);
  return u.v;
}

// Hidden layer: chunk(32x256 f16 in LDS) -> sin(chunk @ W^T + b) -> chunk
__device__ __forceinline__ void mlp_hidden_layer(half_t* chunk,
                                                 const half_t* __restrict__ pw,
                                                 const float* __restrict__ bias,
                                                 int lane) {
  const int mrow  = lane & 15;
  const int hi    = lane >> 4;
  const int kbase = hi * 8;
  // A fragments for both M-tiles (ISA 16-bit A 16x32 layout; 2x ds_load_b128 each)
  v16h a0[8], a1[8];
#pragma unroll
  for (int kt = 0; kt < 8; ++kt) {
    a0[kt] = load_afrag_lds(chunk, mrow,      kt, kbase);
    a1[kt] = load_afrag_lds(chunk, mrow + 16, kt, kbase);
  }
#pragma unroll 4
  for (int t = 0; t < 16; ++t) {
    v8f acc0 = {};
    v8f acc1 = {};
    const half_t* wp = pw + (size_t)(t * 8) * 512 + lane * 16;
#pragma unroll
    for (int kt = 0; kt < 8; ++kt) {
      v16h bfr = *(const v16h*)(wp + (size_t)kt * 512);   // 1 load, 2 WMMAs
      acc0 = wmma_step(a0[kt], bfr, acc0);
      acc1 = wmma_step(a1[kt], bfr, acc1);
    }
    const int col = t * 16 + mrow;
    const float bv = bias[col];
#pragma unroll
    for (int v = 0; v < 8; ++v) {
      int m = v + hi * 8;                                 // C/D layout row
      chunk[m * HID + col]        = (half_t)__sinf(acc0[v] + bv);
      chunk[(m + 16) * HID + col] = (half_t)__sinf(acc1[v] + bv);
    }
  }
}

__global__ __launch_bounds__(256) void siren_mlp_kernel(
    const half_t* __restrict__ feats,          // [totalRows][32] f16
    const half_t* __restrict__ pw0,            // packed layer-0 weights (K=32)
    const half_t* __restrict__ pw1,
    const half_t* __restrict__ pw2,
    const half_t* __restrict__ pw3,
    const half_t* __restrict__ pwf,            // packed final (padded to 16 out)
    const float* __restrict__ b0, const float* __restrict__ b1,
    const float* __restrict__ b2, const float* __restrict__ b3,
    const float* __restrict__ bf,
    float* __restrict__ out, int totalRows) {
  extern __shared__ char smem[];               // 8 waves * 32*256 f16 = 128 KB
  half_t* act = (half_t*)smem;

  const int lane  = threadIdx.x & 31;
  const int wid   = threadIdx.x >> 5;
  const int mrow  = lane & 15;
  const int hi    = lane >> 4;
  const int kbase = hi * 8;
  half_t* chunk = act + wid * (ROWS_PER_WAVE * HID);
  const int row0 = blockIdx.x * ROWS_PER_BLOCK + wid * ROWS_PER_WAVE;

  // Warm L2/WGP$ for the first hidden layer while layer 0 runs (global_prefetch_b8)
  __builtin_prefetch(pw1 + lane * 1024, 0, 1);

  // ---- Layer 0: A (2 x 16x32) straight from global feats ----
  {
    int ar0 = row0 + mrow;       if (ar0 >= totalRows) ar0 = totalRows - 1;
    int ar1 = row0 + 16 + mrow;  if (ar1 >= totalRows) ar1 = totalRows - 1;
    const half_t* fp0 = feats + (size_t)ar0 * C_IN;
    const half_t* fp1 = feats + (size_t)ar1 * C_IN;
    V16 u0, u1;
    u0.h[0] = *(const v8h*)(fp0 + kbase);
    u0.h[1] = *(const v8h*)(fp0 + 16 + kbase);
    u1.h[0] = *(const v8h*)(fp1 + kbase);
    u1.h[1] = *(const v8h*)(fp1 + 16 + kbase);
    v16h a0 = u0.v, a1 = u1.v;
#pragma unroll 4
    for (int t = 0; t < 16; ++t) {
      v16h bfr = *(const v16h*)(pw0 + (size_t)t * 512 + lane * 16);
      v8f acc0 = {};
      v8f acc1 = {};
      acc0 = wmma_step(a0, bfr, acc0);
      acc1 = wmma_step(a1, bfr, acc1);
      const int col = t * 16 + mrow;
      const float bv = b0[col];
#pragma unroll
      for (int v = 0; v < 8; ++v) {
        int m = v + hi * 8;
        chunk[m * HID + col]        = (half_t)__sinf(acc0[v] + bv);
        chunk[(m + 16) * HID + col] = (half_t)__sinf(acc1[v] + bv);
      }
    }
  }
  __syncthreads();

  // ---- Hidden layers 1..3 ----
  __builtin_prefetch(pw2 + lane * 1024, 0, 1);
  mlp_hidden_layer(chunk, pw1, b1, lane);
  __syncthreads();
  __builtin_prefetch(pw3 + lane * 1024, 0, 1);
  mlp_hidden_layer(chunk, pw2, b2, lane);
  __syncthreads();
  __builtin_prefetch(pwf + lane * 128, 0, 1);
  mlp_hidden_layer(chunk, pw3, b3, lane);
  __syncthreads();

  // ---- Final layer: 256 -> 4 (padded to 16), sigmoid on rgb ----
  {
    v8f acc0 = {};
    v8f acc1 = {};
#pragma unroll
    for (int kt = 0; kt < 8; ++kt) {
      v16h a0 = load_afrag_lds(chunk, mrow,      kt, kbase);
      v16h a1 = load_afrag_lds(chunk, mrow + 16, kt, kbase);
      v16h bfr = *(const v16h*)(pwf + (size_t)kt * 512 + lane * 16);
      acc0 = wmma_step(a0, bfr, acc0);
      acc1 = wmma_step(a1, bfr, acc1);
    }
    const int col = mrow;
    if (col < 4) {
      const float bv = bf[col];
#pragma unroll
      for (int v = 0; v < 8; ++v) {
        int m = v + hi * 8;
        int r0 = row0 + m;
        int r1 = row0 + 16 + m;
        if (r0 < totalRows) {
          float x = acc0[v] + bv;
          out[(size_t)r0 * 4 + col] = (col < 3) ? (1.0f / (1.0f + __expf(-x))) : x;
        }
        if (r1 < totalRows) {
          float x = acc1[v] + bv;
          out[(size_t)r1 * 4 + col] = (col < 3) ? (1.0f / (1.0f + __expf(-x))) : x;
        }
      }
    }
  }
}

// ---------------------------------------------------------------------------
// Host launcher
// ---------------------------------------------------------------------------
extern "C" void kernel_launch(void* const* d_in, const int* in_sizes, int n_in,
                              void* d_out, int out_size, void* d_ws, size_t ws_size,
                              hipStream_t stream) {
  const float* points = (const float*)d_in[0];
  const float* vol    = (const float*)d_in[1];
  const float* W0 = (const float*)d_in[2];  const float* b0 = (const float*)d_in[3];
  const float* W1 = (const float*)d_in[4];  const float* b1 = (const float*)d_in[5];
  const float* W2 = (const float*)d_in[6];  const float* b2 = (const float*)d_in[7];
  const float* W3 = (const float*)d_in[8];  const float* b3 = (const float*)d_in[9];
  const float* Wf = (const float*)d_in[10]; const float* bf = (const float*)d_in[11];
  float* out = (float*)d_out;

  const int B    = in_sizes[1] / (C_IN * RES * RES * RES);
  const int Nper = in_sizes[0] / (B * 3);
  const int total = B * Nper;                 // 196608

  // Workspace layout (halfs): packed weights then f16 features
  half_t* pw = (half_t*)d_ws;
  const size_t off0 = 0;                      // 16*1*512  = 8192
  const size_t off1 = 8192;                   // 16*8*512  = 65536
  const size_t off2 = 8192 + 65536;
  const size_t off3 = 8192 + 2 * 65536;
  const size_t offf = 8192 + 3 * 65536;       // 1*8*512   = 4096
  half_t* feats = pw + (8192 + 3 * 65536 + 4096);

  // Pack weights into WMMA B-fragment layout
  pack_w_kernel<<<(16 * 1 * 512 + 255) / 256, 256, 0, stream>>>(W0, pw + off0, HID, C_IN, 16, 1);
  pack_w_kernel<<<(16 * 8 * 512 + 255) / 256, 256, 0, stream>>>(W1, pw + off1, HID, HID, 16, 8);
  pack_w_kernel<<<(16 * 8 * 512 + 255) / 256, 256, 0, stream>>>(W2, pw + off2, HID, HID, 16, 8);
  pack_w_kernel<<<(16 * 8 * 512 + 255) / 256, 256, 0, stream>>>(W3, pw + off3, HID, HID, 16, 8);
  pack_w_kernel<<<( 1 * 8 * 512 + 255) / 256, 256, 0, stream>>>(Wf, pw + offf, 4,   HID,  1, 8);

  // Grid sample -> f16 features
  grid_sample_kernel<<<(total + 255) / 256, 256, 0, stream>>>(points, vol, feats, total, Nper);

  // Fused WMMA MLP (8 waves/block, 256 points/block, 128 KB dynamic LDS)
  const int nblocks = (total + ROWS_PER_BLOCK - 1) / ROWS_PER_BLOCK;
  siren_mlp_kernel<<<nblocks, 256, 128 * 1024, stream>>>(
      feats, pw + off0, pw + off1, pw + off2, pw + off3, pw + offf,
      b0, b1, b2, b3, bf, out, total);
}
